// Model_82832739271250
// MI455X (gfx1250) — matrix operand order repaired and checked by reading, compiled, not run on hardware
//
#include <hip/hip_runtime.h>
#include <hip/hip_bf16.h>

#define EMBD 128
#define NV 100000
#define NP 100
#define NC 512
#define EPSV 1e-8f
#define MT 2              // M-tiles (16 rows each) per block
#define ROWS_PB (16 * MT) // 32 rows per block
#define APAD 136          // padded bf16 row stride in LDS (272B, breaks 256B bank period)

typedef __attribute__((ext_vector_type(16))) __bf16 v16bf;
typedef __attribute__((ext_vector_type(8)))  __bf16 v8bf;
typedef __attribute__((ext_vector_type(8)))  float  v8f;

#if __has_builtin(__builtin_amdgcn_tensor_load_to_lds) && __has_builtin(__builtin_amdgcn_s_wait_tensorcnt)
#define HAVE_TDM 1
#if __has_include(<hip/amd_detail/amd_gfx1250_TDM.h>)
#define TDM_SIX_ARGS 1
#endif
#endif

#ifdef HAVE_TDM
typedef unsigned int u32x4 __attribute__((ext_vector_type(4)));
typedef int          i32x8 __attribute__((ext_vector_type(8)));
typedef int          i32x4 __attribute__((ext_vector_type(4)));

// Issue one TDM descriptor: load a [tile_rows x 128] fp32 tile (row stride 128)
// from global into LDS at byte offset lds_byte_off. Rows beyond rows_avail read 0.
// D# bit layout per CDNA5 ISA ch.8 (group0: count/lds_addr/global_addr/type=2;
// group1: data_size=2(4B), tensor_dim0=128, tile_dim0=128, tile_dim1=tile_rows,
// tensor_dim1=rows_avail, tensor_dim0_stride=128).
__device__ __forceinline__ void tdm_load_tile_f32(unsigned lds_byte_off, const float* gaddr,
                                                  int rows_avail, int tile_rows) {
    unsigned long long ga = (unsigned long long)(uintptr_t)gaddr;
    u32x4 g0 = { 1u,                                   // count=1 (valid user descriptor)
                 lds_byte_off,                         // lds_addr [63:32]
                 (unsigned)(ga & 0xFFFFFFFFu),         // global_addr lo
                 (unsigned)((ga >> 32) & 0x1FFFFFFull) | 0x80000000u }; // addr hi | type=2
    unsigned rt = (unsigned)rows_avail;
    i32x8 g1 = { (int)0x00020000u,                     // data_size=2 -> 4 bytes
                 (int)(128u << 16),                    // tensor_dim0[15:0] in bits[63:48]
                 (int)((rt & 0xFFFFu) << 16),          // dim0 hi=0 | tensor_dim1 lo
                 (int)((rt >> 16) | (128u << 16)),     // tensor_dim1 hi | tile_dim0=128
                 (int)(unsigned)tile_rows,             // tile_dim1 (tile_dim2=0)
                 128,                                  // tensor_dim0_stride lo
                 0, 0 };
    i32x4 z4 = { 0, 0, 0, 0 };
#ifdef TDM_SIX_ARGS
    i32x8 z8 = { 0, 0, 0, 0, 0, 0, 0, 0 };
    __builtin_amdgcn_tensor_load_to_lds(g0, g1, z4, z4, z8, 0);
#else
    __builtin_amdgcn_tensor_load_to_lds(g0, g1, z4, z4, 0);
#endif
}
#endif

// ---------------------------------------------------------------- utilities

__global__ void zero_f32(float* p, long n) {
    long stride = (long)gridDim.x * blockDim.x;
    for (long i = (long)blockIdx.x * blockDim.x + threadIdx.x; i < n; i += stride)
        p[i] = 0.0f;
}

// S[r] = sum_d emb[r, d]
__global__ void rowsum_kernel(const float* emb, float* S, int R) {
    int r = blockIdx.x * blockDim.x + threadIdx.x;
    if (r >= R) return;
    const float4* row = (const float4*)(emb + (long)r * EMBD);
    float acc = 0.f;
#pragma unroll
    for (int i = 0; i < EMBD / 4; ++i) {
        float4 x = row[i];
        acc += x.x + x.y + x.z + x.w;
    }
    S[r] = acc;
}

// ------------------------------------------------- weight precompute (bf16)

// Ut layout: [g][n][k]; U1 = Wa[0:128], U2 = Wa[128:256]+W1, U3 = Wa[256:384]+W2
__global__ void prep_U(const float* Wa, const float* W1, const float* W2, __bf16* Ut) {
    int idx = blockIdx.x * blockDim.x + threadIdx.x;
    if (idx >= 3 * EMBD * EMBD) return;
    int k = idx & 127;
    int n = (idx >> 7) & 127;
    int g = idx >> 14;
    float w = Wa[((g << 7) + k) * EMBD + n];
    if (g == 1) w += W1[k * EMBD + n];
    if (g == 2) w += W2[k * EMBD + n];
    Ut[idx] = (__bf16)w;
}

__global__ void prep_bias(const float* ba, const float* b1, const float* b2, float* c) {
    int i = threadIdx.x;
    if (i < EMBD) c[i] = ba[i] + b1[i] + b2[i];
}

// ---------------------------------------------------------------- SPMM (COO)

__global__ void spmm_atomic(const int* __restrict__ row, const int* __restrict__ col,
                            const float* __restrict__ val, const float* __restrict__ emb,
                            float* __restrict__ out, int nnz) {
    int e = (blockIdx.x * blockDim.x + threadIdx.x) >> 5;
    if (e >= nnz) return;
    int lane = threadIdx.x & 31;
    int r = row[e], c = col[e];
    float v = val[e];
    const float4* src = (const float4*)(emb + (long)c * EMBD);
    float4 x = src[lane];
    float* dst = out + (long)r * EMBD + lane * 4;
    atomicAdd(dst + 0, v * x.x);
    atomicAdd(dst + 1, v * x.y);
    atomicAdd(dst + 2, v * x.z);
    atomicAdd(dst + 3, v * x.w);
}

// ---------------------------------------------------------------- intra_gate

__global__ void intra_Z(const float* __restrict__ mat, const float* __restrict__ S,
                        float* __restrict__ Z, int C) {
    __shared__ float red[256];
    int i = blockIdx.y;
    int j = blockIdx.x * blockDim.x + threadIdx.x;
    float m = mat[i];
    float a = (j < C) ? __expf(m * S[j]) : 0.f;
    red[threadIdx.x] = a;
    __syncthreads();
    for (int s = 128; s > 0; s >>= 1) {
        if (threadIdx.x < s) red[threadIdx.x] += red[threadIdx.x + s];
        __syncthreads();
    }
    if (threadIdx.x == 0) atomicAdd(&Z[i], red[0]);
}

__global__ void intra_edge(const int* __restrict__ row, const int* __restrict__ col,
                           const float* __restrict__ val, const float* __restrict__ mat,
                           const float* __restrict__ S, const float* __restrict__ emb2,
                           float* __restrict__ num, float* __restrict__ Wsum, int nnz) {
    int e = (blockIdx.x * blockDim.x + threadIdx.x) >> 5;
    if (e >= nnz) return;
    int lane = threadIdx.x & 31;
    int r = row[e], c = col[e];
    float w = __expf(mat[r] * S[c]) * val[e];
    if (lane == 0) atomicAdd(&Wsum[r], w);
    const float4* src = (const float4*)(emb2 + (long)c * EMBD);
    float4 x = src[lane];
    float* dst = num + (long)r * EMBD + lane * 4;
    atomicAdd(dst + 0, w * x.x);
    atomicAdd(dst + 1, w * x.y);
    atomicAdd(dst + 2, w * x.z);
    atomicAdd(dst + 3, w * x.w);
}

__global__ void intra_scale(float* __restrict__ num, const float* __restrict__ Wsum,
                            const float* __restrict__ Z, int R) {
    int idx = blockIdx.x * blockDim.x + threadIdx.x;
    if (idx >= R * EMBD) return;
    int i = idx >> 7;
    num[idx] *= 1.0f / (Wsum[i] + EPSV * Z[i]);
}

// -------------------------------------------- fused WMMA GEMM + gate combine
//
// gpre = e1@U1 + e2@U2 + e3@U3 + cbias ; g = sigmoid(gpre)
// out  = e1 + g*e2 + (1-g)*e3
//
// Block = 256 threads = 8 wave32s; block owns 32 rows (2 M-tiles), wave w owns
// N-tile [16w,16w+16). A is staged once per block into LDS as bf16 (TDM path
// stages fp32 via tensor_load_to_lds, then converts). Each wave: 24 WMMAs,
// with every B fragment reused across both M-tiles.
__global__ void __launch_bounds__(256)
gate_gemm(const float* __restrict__ e1, const float* __restrict__ e2,
          const float* __restrict__ e3, const __bf16* __restrict__ Ut,
          const float* __restrict__ cbias, float* __restrict__ out, int M) {
    __shared__ __bf16 Asbf[3 * ROWS_PB * APAD];     // 26112 B
#ifdef HAVE_TDM
    __shared__ float  As32[ROWS_PB * EMBD];         // 16384 B staging, reused per g
#endif

    int wave  = threadIdx.x >> 5;
    int lane  = threadIdx.x & 31;
    int row16 = lane & 15;
    int half  = lane >> 4;
    long mbase = (long)blockIdx.x * ROWS_PB;

    const float* Emat[3] = { e1, e2, e3 };

    // ---- stage A tiles (3 matrices x 32 rows x 128 cols) into LDS as bf16
#ifdef HAVE_TDM
    {
        int rows_avail = (int)((long)M - mbase);    // >0 by grid construction
        unsigned lds_off = (unsigned)(uintptr_t)(const void*)&As32[0];
        for (int g = 0; g < 3; ++g) {
            if (threadIdx.x < 32) {                 // one wave issues the TDM op
                tdm_load_tile_f32(lds_off, Emat[g] + mbase * EMBD, rows_avail, ROWS_PB);
                __builtin_amdgcn_s_wait_tensorcnt(0);
            }
            __syncthreads();
            for (int t = threadIdx.x; t < ROWS_PB * EMBD; t += 256) {
                int r = t >> 7, k = t & 127;
                Asbf[(g * ROWS_PB + r) * APAD + k] = (__bf16)As32[t];
            }
            __syncthreads();
        }
    }
#else
    for (int t = threadIdx.x; t < 3 * ROWS_PB * EMBD; t += 256) {
        int g = t >> 12;
        int r = (t >> 7) & (ROWS_PB - 1);
        int k = t & 127;
        long row = mbase + r;
        float v = (row < M) ? Emat[g][row * EMBD + k] : 0.f;
        Asbf[(g * ROWS_PB + r) * APAD + k] = (__bf16)v;
    }
    __syncthreads();
#endif

    int ncol = wave * 16 + row16;
    v8f acc0 = {};
    v8f acc1 = {};
#pragma unroll
    for (int g = 0; g < 3; ++g) {
        const __bf16* Brow = Ut + (long)g * EMBD * EMBD + (long)ncol * EMBD;
#pragma unroll
        for (int k0 = 0; k0 < EMBD; k0 += 32) {
            // B fragment: lane holds column ncol; lanes 0-15: K k0..k0+15,
            // lanes 16-31: K k0+16..k0+31 -> 32B contiguous per lane
            v16bf b = *(const v16bf*)(Brow + k0 + half * 16);

            // A fragments from LDS (contiguous k order, not the ISA table order):
            // lanes 0-15: K k0+0..15; lanes 16-31: K k0+16..31 (matches B)
            int kbA = k0 + half * 16;
#pragma unroll
            for (int t = 0; t < MT; ++t) {
                const __bf16* ap = &Asbf[(g * ROWS_PB + t * 16 + row16) * APAD];
                v8bf lo = *(const v8bf*)(ap + kbA);
                v8bf hi = *(const v8bf*)(ap + kbA + 8);
                v16bf a;
#pragma unroll
                for (int j = 0; j < 8; ++j) { a[j] = lo[j]; a[j + 8] = hi[j]; }
                if (t == 0)
                    acc0 = __builtin_amdgcn_wmma_f32_16x16x32_bf16(
                        false, a, false, b, (short)0, acc0, false, false);
                else
                    acc1 = __builtin_amdgcn_wmma_f32_16x16x32_bf16(
                        false, a, false, b, (short)0, acc1, false, false);
            }
        }
    }

    // epilogue: C/D layout: VGPR r, lanes 0-15 -> (M=r,N=lane), 16-31 -> (M=r+8)
#pragma unroll
    for (int t = 0; t < MT; ++t) {
        const v8f& acc = (t == 0) ? acc0 : acc1;
#pragma unroll
        for (int r = 0; r < 8; ++r) {
            long orow = mbase + t * 16 + r + half * 8;
            int ocol = wave * 16 + row16;
            if (orow < M) {
                float gp = acc[r] + cbias[ocol];
                float gg = 1.0f / (1.0f + __expf(-gp));
                long off = orow * EMBD + ocol;
                out[off] = e1[off] + gg * e2[off] + (1.0f - gg) * e3[off];
            }
        }
    }
}

// ------------------------------------------------------------------- launch

extern "C" void kernel_launch(void* const* d_in, const int* in_sizes, int n_in,
                              void* d_out, int out_size, void* d_ws, size_t ws_size,
                              hipStream_t stream) {
    (void)n_in; (void)out_size; (void)ws_size;

    const float* emb_v = (const float*)d_in[0];
    const float* emb_p = (const float*)d_in[1];
    const float* emb_c = (const float*)d_in[2];

    const int *vp_row = (const int*)d_in[3],  *vp_col = (const int*)d_in[4];
    const float *vp_val = (const float*)d_in[5];
    const int *vc_row = (const int*)d_in[6],  *vc_col = (const int*)d_in[7];
    const float *vc_val = (const float*)d_in[8];
    const int *pv_row = (const int*)d_in[9],  *pv_col = (const int*)d_in[10];
    const float *pv_val = (const float*)d_in[11];
    const int *pc_row = (const int*)d_in[12], *pc_col = (const int*)d_in[13];
    const float *pc_val = (const float*)d_in[14];
    const int *cp_row = (const int*)d_in[15], *cp_col = (const int*)d_in[16];
    const float *cp_val = (const float*)d_in[17];
    const int *cv_row = (const int*)d_in[18], *cv_col = (const int*)d_in[19];
    const float *cv_val = (const float*)d_in[20];

    int vp_nnz = in_sizes[3], vc_nnz = in_sizes[6], pv_nnz = in_sizes[9];
    int pc_nnz = in_sizes[12], cp_nnz = in_sizes[15], cv_nnz = in_sizes[18];

    const float* mat_pv = (const float*)d_in[21];
    const float* mat_pc = (const float*)d_in[22];
    const float* mat_cp = (const float*)d_in[23];
    const float* mat_cv = (const float*)d_in[24];

    const float* Wai = (const float*)d_in[25]; const float* bai = (const float*)d_in[26];
    const float* Wi1 = (const float*)d_in[27]; const float* bi1 = (const float*)d_in[28];
    const float* Wi2 = (const float*)d_in[29]; const float* bi2 = (const float*)d_in[30];
    const float* Wap = (const float*)d_in[31]; const float* bap = (const float*)d_in[32];
    const float* Wp1 = (const float*)d_in[33]; const float* bp1 = (const float*)d_in[34];
    const float* Wp2 = (const float*)d_in[35]; const float* bp2 = (const float*)d_in[36];
    const float* Wac = (const float*)d_in[37]; const float* bac = (const float*)d_in[38];
    const float* Wc1 = (const float*)d_in[39]; const float* bc1 = (const float*)d_in[40];
    const float* Wc2 = (const float*)d_in[41]; const float* bc2 = (const float*)d_in[42];

    // ---- workspace bump allocator (256B aligned)
    char* wp = (char*)d_ws;
    auto alloc = [&](size_t bytes) -> void* {
        void* r = (void*)wp;
        wp += (bytes + 255) & ~(size_t)255;
        return r;
    };

    char* zstart = wp;
    float* agg_vp = (float*)alloc((size_t)NV * EMBD * 4);
    float* agg_vc = (float*)alloc((size_t)NV * EMBD * 4);
    float* num_pv = (float*)alloc((size_t)NP * EMBD * 4);
    float* num_pc = (float*)alloc((size_t)NP * EMBD * 4);
    float* num_cp = (float*)alloc((size_t)NC * EMBD * 4);
    float* num_cv = (float*)alloc((size_t)NC * EMBD * 4);
    float* Wpv = (float*)alloc(NP * 4);  float* Wpc = (float*)alloc(NP * 4);
    float* Wcp = (float*)alloc(NC * 4);  float* Wcv = (float*)alloc(NC * 4);
    float* Zpv = (float*)alloc(NP * 4);  float* Zpc = (float*)alloc(NP * 4);
    float* Zcp = (float*)alloc(NC * 4);  float* Zcv = (float*)alloc(NC * 4);
    long zero_count = (long)(wp - zstart) / 4;

    float* Sv = (float*)alloc((size_t)NV * 4);
    float* Sp = (float*)alloc((size_t)NP * 4);
    float* Sc = (float*)alloc((size_t)NC * 4);

    float* ev1 = (float*)alloc((size_t)NV * EMBD * 4);
    float* ep1 = (float*)alloc((size_t)NP * EMBD * 4);
    float* ec1 = (float*)alloc((size_t)NC * EMBD * 4);

    __bf16* Ui = (__bf16*)alloc((size_t)3 * EMBD * EMBD * 2);
    __bf16* Up = (__bf16*)alloc((size_t)3 * EMBD * EMBD * 2);
    __bf16* Uc = (__bf16*)alloc((size_t)3 * EMBD * EMBD * 2);
    float* ci  = (float*)alloc(EMBD * 4);
    float* cpb = (float*)alloc(EMBD * 4);
    float* ccb = (float*)alloc(EMBD * 4);

    int ublk = (3 * EMBD * EMBD + 255) / 256;
    prep_U<<<ublk, 256, 0, stream>>>(Wai, Wi1, Wi2, Ui);
    prep_U<<<ublk, 256, 0, stream>>>(Wap, Wp1, Wp2, Up);
    prep_U<<<ublk, 256, 0, stream>>>(Wac, Wc1, Wc2, Uc);
    prep_bias<<<1, EMBD, 0, stream>>>(bai, bi1, bi2, ci);
    prep_bias<<<1, EMBD, 0, stream>>>(bap, bp1, bp2, cpb);
    prep_bias<<<1, EMBD, 0, stream>>>(bac, bc1, bc2, ccb);

    auto run_layer = [&](const float* ev, const float* ep, const float* ec,
                         float* ov, float* op, float* oc) {
        zero_f32<<<4096, 256, 0, stream>>>((float*)zstart, zero_count);
        rowsum_kernel<<<(NV + 255) / 256, 256, 0, stream>>>(ev, Sv, NV);
        rowsum_kernel<<<(NP + 255) / 256, 256, 0, stream>>>(ep, Sp, NP);
        rowsum_kernel<<<(NC + 255) / 256, 256, 0, stream>>>(ec, Sc, NC);
        spmm_atomic<<<(vp_nnz * 32 + 255) / 256, 256, 0, stream>>>(
            vp_row, vp_col, vp_val, ep, agg_vp, vp_nnz);
        spmm_atomic<<<(vc_nnz * 32 + 255) / 256, 256, 0, stream>>>(
            vc_row, vc_col, vc_val, ec, agg_vc, vc_nnz);
        intra_Z<<<dim3((NV + 255) / 256, NP), 256, 0, stream>>>(mat_pv, Sv, Zpv, NV);
        intra_Z<<<dim3((NC + 255) / 256, NP), 256, 0, stream>>>(mat_pc, Sc, Zpc, NC);
        intra_Z<<<dim3((NP + 255) / 256, NC), 256, 0, stream>>>(mat_cp, Sp, Zcp, NP);
        intra_Z<<<dim3((NV + 255) / 256, NC), 256, 0, stream>>>(mat_cv, Sv, Zcv, NV);

        intra_edge<<<(pv_nnz * 32 + 255) / 256, 256, 0, stream>>>(
            pv_row, pv_col, pv_val, mat_pv, Sv, ev, num_pv, Wpv, pv_nnz);
        intra_edge<<<(pc_nnz * 32 + 255) / 256, 256, 0, stream>>>(
            pc_row, pc_col, pc_val, mat_pc, Sc, ec, num_pc, Wpc, pc_nnz);
        intra_edge<<<(cp_nnz * 32 + 255) / 256, 256, 0, stream>>>(
            cp_row, cp_col, cp_val, mat_cp, Sp, ep, num_cp, Wcp, cp_nnz);
        intra_edge<<<(cv_nnz * 32 + 255) / 256, 256, 0, stream>>>(
            cv_row, cv_col, cv_val, mat_cv, Sv, ev, num_cv, Wcv, cv_nnz);

        intra_scale<<<(NP * EMBD + 255) / 256, 256, 0, stream>>>(num_pv, Wpv, Zpv, NP);
        intra_scale<<<(NP * EMBD + 255) / 256, 256, 0, stream>>>(num_pc, Wpc, Zpc, NP);
        intra_scale<<<(NC * EMBD + 255) / 256, 256, 0, stream>>>(num_cp, Wcp, Zcp, NC);
        intra_scale<<<(NC * EMBD + 255) / 256, 256, 0, stream>>>(num_cv, Wcv, Zcv, NC);

        gate_gemm<<<(NV + ROWS_PB - 1) / ROWS_PB, 256, 0, stream>>>(
            ev, agg_vp, agg_vc, Ui, ci, ov, NV);
        gate_gemm<<<(NP + ROWS_PB - 1) / ROWS_PB, 256, 0, stream>>>(
            ep, num_pv, num_pc, Up, cpb, op, NP);
        gate_gemm<<<(NC + ROWS_PB - 1) / ROWS_PB, 256, 0, stream>>>(
            ec, num_cp, num_cv, Uc, ccb, oc, NC);
    };

    float* out_v = (float*)d_out;
    float* out_p = out_v + (size_t)NV * EMBD;
    float* out_c = out_p + (size_t)NP * EMBD;

    run_layer(emb_v, emb_p, emb_c, ev1, ep1, ec1);
    run_layer(ev1, ep1, ec1, out_v, out_p, out_c);
}
